// WDNLeakGNN_4080218931515
// MI455X (gfx1250) — compile-verified, matrix operand order checked by reading, not run on hardware
//
#include <hip/hip_runtime.h>
#include <hip/hip_bf16.h>

#define N_NODES 200000
#define N_EDGES 600000
#define HID 128

typedef __attribute__((ext_vector_type(16))) __bf16 v16bf;
typedef __attribute__((ext_vector_type(8)))  float  v8f;
typedef __attribute__((ext_vector_type(4)))  unsigned int u32x4;
typedef __attribute__((ext_vector_type(8)))  int  i32x8;
typedef __attribute__((ext_vector_type(4)))  int  i32x4;

static __device__ __forceinline__ v8f wmma_bf16(v16bf a, v16bf b, v8f c) {
  return __builtin_amdgcn_wmma_f32_16x16x32_bf16(false, a, false, b, (short)0, c,
                                                 false, false);
}

// ---------------------------------------------------------------------------
// TDM: 1D contiguous copy of n fp32 elements from global -> LDS.
// D# per CDNA5 ISA:
//  group0: count=1 | lds_addr | global_addr[56:0] | type=2
//  group1: data_size=4B, tensor_dim0=n, tensor_dim1=1, tile_dim0=n,
//          tile_dim1=1, tensor_dim0_stride=n, no pad/iterate/multicast
// Issued once per wave (EXEC ignored); completion via TENSORcnt.
// 6-arg builtin form (amdgpu-toolchain / clang-23).
// ---------------------------------------------------------------------------
static __device__ __forceinline__ void tdm_load_1d(unsigned lds_off,
                                                   const void* gptr,
                                                   unsigned n) {
  unsigned long long ga = (unsigned long long)(size_t)gptr;
  u32x4 g0 = {1u,                                   // count=1, user mode
              lds_off,                              // lds_addr
              (unsigned)ga,                         // global_addr[31:0]
              (unsigned)((ga >> 32) & 0x01ffffffu) | 0x80000000u}; // [56:32]|type=2
  i32x8 g1 = {(int)0x00020000,                      // data_size = 4 bytes
              (int)(n << 16),                       // tensor_dim0[15:0]
              (int)((n >> 16) | (1u << 16)),        // tensor_dim0[31:16]|tensor_dim1=1
              (int)(n << 16),                       // tile_dim0 = n
              1,                                    // tile_dim1 = 1
              (int)n,                               // tensor_dim0_stride = n
              0, 0};
  i32x4 gz4 = {0, 0, 0, 0};
  i32x8 gz8 = {0, 0, 0, 0, 0, 0, 0, 0};
  __builtin_amdgcn_tensor_load_to_lds(g0, g1, gz4, gz4, gz8, 0);
}

static __device__ __forceinline__ unsigned lds_offset(const void* p) {
  return (unsigned)(size_t)p;  // LDS aperture: low 32 bits = wave-relative offset
}

// B fragments pre-swizzled as [kt][nt(8)][lane(32)][j(16)] contiguous per lane.
static __device__ __forceinline__ v16bf load_b_frag(const __bf16* __restrict__ wf,
                                                    int kt, int nt, int lane) {
  const __bf16* p = wf + ((((size_t)kt * 8 + nt) * 32 + lane) * 16);
  v16bf b;
#pragma unroll
  for (int j = 0; j < 16; ++j) b[j] = p[j];
  return b;
}

// ---------------------------------------------------------------------------
// Swizzle fp32 weight [K][128] into bf16 B-fragment layout (zero-pad k>=ksrc).
// ---------------------------------------------------------------------------
__global__ void swizzle_b_kernel(const float* __restrict__ w, __bf16* __restrict__ out,
                                 int ksrc, int ktiles) {
  int total = ktiles * 8 * 32 * 16;
  int idx = blockIdx.x * blockDim.x + threadIdx.x;
  if (idx >= total) return;
  int j    = idx & 15;
  int lane = (idx >> 4) & 31;
  int nt   = (idx >> 9) & 7;
  int kt   = idx >> 12;
  int kloc = (j < 8) ? ((lane >> 4) * 8 + j) : (16 + (lane >> 4) * 8 + (j - 8));
  int k = kt * 32 + kloc;
  int n = nt * 16 + (lane & 15);
  out[idx] = (k < ksrc) ? (__bf16)w[(size_t)k * HID + n] : (__bf16)0.0f;
}

__global__ void zero_kernel(float* __restrict__ p, size_t n) {
  size_t i = (size_t)blockIdx.x * blockDim.x + threadIdx.x;
  if (i < n) p[i] = 0.0f;
}

// ---------------------------------------------------------------------------
// Encoder GEMM: out[nrows][128] = in[nrows][ld](:, :ksrc) @ W + bias
// Block = 16 rows (contiguous 16*ld fp32) staged into LDS by one TDM op.
// ---------------------------------------------------------------------------
__global__ void enc_gemm_kernel(const float* __restrict__ in, int ld, int ksrc,
                                const __bf16* __restrict__ wf,
                                const float* __restrict__ bias,
                                float* __restrict__ out) {
  __shared__ float sh_in[16 * 32];
  int lane = threadIdx.x & 31;
  int wave = threadIdx.x >> 5;
  int m0 = blockIdx.x << 4;
  int m = lane & 15;
  int kb = (lane >> 4) << 3;

  if (wave == 0) {
    tdm_load_1d(lds_offset(sh_in), in + (size_t)m0 * ld, (unsigned)(16 * ld));
    __builtin_amdgcn_s_wait_tensorcnt(0);
  }
  __syncthreads();

  const float* rowp = sh_in + m * ld;
  v16bf a;
#pragma unroll
  for (int j = 0; j < 8; ++j) {
    int c0 = kb + j, c1 = 16 + kb + j;
    a[j]     = (c0 < ksrc) ? (__bf16)rowp[c0] : (__bf16)0.0f;
    a[8 + j] = (c1 < ksrc) ? (__bf16)rowp[c1] : (__bf16)0.0f;
  }
#pragma unroll
  for (int t = 0; t < 2; ++t) {
    int nt = wave * 2 + t;
    v8f acc = {};
    acc = wmma_bf16(a, load_b_frag(wf, 0, nt, lane), acc);
#pragma unroll
    for (int v = 0; v < 8; ++v) {
      int r = v + ((lane >> 4) << 3);
      int col = (nt << 4) + m;
      out[(size_t)(m0 + r) * HID + col] = acc[v] + bias[col];
    }
  }
}

// ---------------------------------------------------------------------------
// Message + scatter: agg[dst] += relu(h[src] + e).
// ---------------------------------------------------------------------------
__global__ void message_kernel(const float* __restrict__ h, const float* __restrict__ e,
                               const int* __restrict__ ei, float* __restrict__ agg) {
  int tid = blockIdx.x * blockDim.x + threadIdx.x;
  int edge = tid >> 5;
  if (edge >= N_EDGES) return;
  int c0 = (tid & 31) << 2;
  int s = ei[edge];
  int d = ei[N_EDGES + edge];
  float4 hv = *(const float4*)(h + (size_t)s * HID + c0);
  float4 ev = *(const float4*)(e + (size_t)edge * HID + c0);
  float4 mv;
  mv.x = fmaxf(hv.x + ev.x, 0.0f);
  mv.y = fmaxf(hv.y + ev.y, 0.0f);
  mv.z = fmaxf(hv.z + ev.z, 0.0f);
  mv.w = fmaxf(hv.w + ev.w, 0.0f);
  float* ap = agg + (size_t)d * HID + c0;
  atomicAdd(ap + 0, mv.x);
  atomicAdd(ap + 1, mv.y);
  atomicAdd(ap + 2, mv.z);
  atomicAdd(ap + 3, mv.w);
}

// ---------------------------------------------------------------------------
// Fused GINE node MLP:
//   z = relu(((1+eps)*h + agg) @ W1 + b1) @ W2 + b2, plus LN (sum,sumsq).
// h/agg 16-row tiles (8 KB each, contiguous) staged into LDS via TDM,
// eliminating the 4x redundant per-wave global fragment loads.
// ---------------------------------------------------------------------------
__global__ void gine_mlp_kernel(const float* __restrict__ h, const float* __restrict__ agg,
                                const float* __restrict__ epsp, int layer,
                                const __bf16* __restrict__ w1f, const float* __restrict__ b1,
                                const __bf16* __restrict__ w2f, const float* __restrict__ b2,
                                float* __restrict__ z, float* __restrict__ stats) {
  __shared__ float sh_h[16 * HID];
  __shared__ float sh_g[16 * HID];
  __shared__ __bf16 t[16][HID];
  __shared__ float red[2];
  int lane = threadIdx.x & 31;
  int wave = threadIdx.x >> 5;
  int m0 = blockIdx.x << 4;
  int m = lane & 15;
  int kb = (lane >> 4) << 3;
  float c1 = 1.0f + epsp[layer];

  if (threadIdx.x < 2) red[threadIdx.x] = 0.0f;
  if (wave == 0) {
    tdm_load_1d(lds_offset(sh_h), h + (size_t)m0 * HID, 16 * HID);
    tdm_load_1d(lds_offset(sh_g), agg + (size_t)m0 * HID, 16 * HID);
    __builtin_amdgcn_s_wait_tensorcnt(0);
  }
  __syncthreads();

  const float* hrow = sh_h + m * HID;
  const float* grow = sh_g + m * HID;

  // A fragments for z_in = (1+eps)*h + agg, K = 128 = 4 k-tiles
  v16bf afr[4];
#pragma unroll
  for (int kt = 0; kt < 4; ++kt) {
    int coff = kt << 5;
#pragma unroll
    for (int j = 0; j < 8; ++j) {
      int ca = coff + kb + j, cb = coff + 16 + kb + j;
      afr[kt][j]     = (__bf16)(c1 * hrow[ca] + grow[ca]);
      afr[kt][8 + j] = (__bf16)(c1 * hrow[cb] + grow[cb]);
    }
  }

  // GEMM1 + bias + relu -> LDS (bf16)
#pragma unroll
  for (int tix = 0; tix < 2; ++tix) {
    int nt = wave * 2 + tix;
    v8f acc = {};
#pragma unroll
    for (int kt = 0; kt < 4; ++kt)
      acc = wmma_bf16(afr[kt], load_b_frag(w1f, kt, nt, lane), acc);
#pragma unroll
    for (int v = 0; v < 8; ++v) {
      int r = v + ((lane >> 4) << 3);
      int col = (nt << 4) + m;
      t[r][col] = (__bf16)fmaxf(acc[v] + b1[col], 0.0f);
    }
  }
  __syncthreads();

  // GEMM2 from LDS, write z, accumulate LN statistics
  float s = 0.0f, s2 = 0.0f;
#pragma unroll
  for (int tix = 0; tix < 2; ++tix) {
    int nt = wave * 2 + tix;
    v8f acc = {};
#pragma unroll
    for (int kt = 0; kt < 4; ++kt) {
      int coff = kt << 5;
      v16bf a;
#pragma unroll
      for (int j = 0; j < 8; ++j) {
        a[j]     = t[m][coff + kb + j];
        a[8 + j] = t[m][coff + 16 + kb + j];
      }
      acc = wmma_bf16(a, load_b_frag(w2f, kt, nt, lane), acc);
    }
#pragma unroll
    for (int v = 0; v < 8; ++v) {
      int r = v + ((lane >> 4) << 3);
      int col = (nt << 4) + m;
      float zv = acc[v] + b2[col];
      z[(size_t)(m0 + r) * HID + col] = zv;
      s += zv;
      s2 += zv * zv;
    }
  }
  atomicAdd(&red[0], s);    // ds_add_f32
  atomicAdd(&red[1], s2);
  __syncthreads();
  if (threadIdx.x == 0) {
    atomicAdd(&stats[0], red[0]);
    atomicAdd(&stats[1], red[1]);
  }
}

// graph-mode LayerNorm + relu: h = relu((z - mu) * rsqrt(var+eps) * g + b)
__global__ void ln_relu_kernel(const float* __restrict__ z, const float* __restrict__ stats,
                               const float* __restrict__ g, const float* __restrict__ b,
                               float* __restrict__ h) {
  const float invn = 1.0f / ((float)N_NODES * (float)HID);
  size_t idx = (size_t)blockIdx.x * blockDim.x + threadIdx.x;
  if (idx >= (size_t)N_NODES * HID) return;
  float mu = stats[0] * invn;
  float var = stats[1] * invn - mu * mu;
  float inv = rsqrtf(var + 1e-5f);
  int c = (int)(idx & (HID - 1));
  float v = (z[idx] - mu) * inv * g[c] + b[c];
  h[idx] = fmaxf(v, 0.0f);
}

// ---------------------------------------------------------------------------
// Fused edge MLP: logits = relu(concat(h[src], h[dst], e) @ W1 + b1) @ w2 + b2
// e rows for the block's 16 edges are contiguous -> staged by TDM; h[src]/h[dst]
// remain random-row global gathers. hmid never hits HBM (LDS ds_add_f32).
// ---------------------------------------------------------------------------
__global__ void edge_mlp_kernel(const float* __restrict__ h, const float* __restrict__ e,
                                const int* __restrict__ ei,
                                const __bf16* __restrict__ w1f, const float* __restrict__ b1,
                                const float* __restrict__ w2, const float* __restrict__ b2p,
                                float* __restrict__ out) {
  __shared__ float sh_e[16 * HID];
  __shared__ float accrow[16];
  int lane = threadIdx.x & 31;
  int wave = threadIdx.x >> 5;
  int e0 = blockIdx.x << 4;
  int m = lane & 15;
  int kb = (lane >> 4) << 3;
  int edge = e0 + m;
  int s = ei[edge];
  int d = ei[N_EDGES + edge];
  const float* ph = h + (size_t)s * HID;
  const float* pd = h + (size_t)d * HID;

  if (wave == 0) {
    tdm_load_1d(lds_offset(sh_e), e + (size_t)e0 * HID, 16 * HID);
    __builtin_amdgcn_s_wait_tensorcnt(0);
  }
  if (threadIdx.x < 16) accrow[threadIdx.x] = 0.0f;
  __syncthreads();

#pragma unroll
  for (int tix = 0; tix < 2; ++tix) {
    int nt = wave * 2 + tix;
    v8f acc = {};
#pragma unroll
    for (int kt = 0; kt < 12; ++kt) {
      int coff = (kt & 3) << 5;
      v16bf a;
      if (kt < 8) {
        const float* base = (kt < 4) ? ph : pd;
#pragma unroll
        for (int j = 0; j < 8; ++j) {
          a[j]     = (__bf16)base[coff + kb + j];
          a[8 + j] = (__bf16)base[coff + 16 + kb + j];
        }
      } else {
#pragma unroll
        for (int j = 0; j < 8; ++j) {
          a[j]     = (__bf16)sh_e[m * HID + coff + kb + j];
          a[8 + j] = (__bf16)sh_e[m * HID + coff + 16 + kb + j];
        }
      }
      acc = wmma_bf16(a, load_b_frag(w1f, kt, nt, lane), acc);
    }
#pragma unroll
    for (int v = 0; v < 8; ++v) {
      int r = v + ((lane >> 4) << 3);
      int col = (nt << 4) + m;
      float hm = fmaxf(acc[v] + b1[col], 0.0f);
      atomicAdd(&accrow[r], hm * w2[col]);  // ds_add_f32
    }
  }
  __syncthreads();
  if (threadIdx.x < 16) out[e0 + threadIdx.x] = accrow[threadIdx.x] + b2p[0];
}

extern "C" void kernel_launch(void* const* d_in, const int* in_sizes, int n_in,
                              void* d_out, int out_size, void* d_ws, size_t ws_size,
                              hipStream_t stream) {
  (void)in_sizes; (void)n_in; (void)out_size; (void)ws_size;
  const float* x        = (const float*)d_in[0];
  const int*   ei       = (const int*)d_in[1];
  const float* eattr    = (const float*)d_in[2];
  const float* enc_w    = (const float*)d_in[3];
  const float* enc_b    = (const float*)d_in[4];
  const float* ee_w     = (const float*)d_in[5];
  const float* ee_b     = (const float*)d_in[6];
  const float* gine_w1  = (const float*)d_in[7];
  const float* gine_b1  = (const float*)d_in[8];
  const float* gine_w2  = (const float*)d_in[9];
  const float* gine_b2  = (const float*)d_in[10];
  const float* gine_eps = (const float*)d_in[11];
  const float* ln_g     = (const float*)d_in[12];
  const float* ln_b     = (const float*)d_in[13];
  const float* mlp_w1   = (const float*)d_in[14];
  const float* mlp_b1   = (const float*)d_in[15];
  const float* mlp_w2   = (const float*)d_in[16];
  const float* mlp_b2   = (const float*)d_in[17];
  float* out = (float*)d_out;

  char* p = (char*)d_ws;
  float* h    = (float*)p; p += (size_t)N_NODES * HID * 4;
  float* z    = (float*)p; p += (size_t)N_NODES * HID * 4;
  float* agg  = (float*)p; p += (size_t)N_NODES * HID * 4;
  float* ebuf = (float*)p; p += (size_t)N_EDGES * HID * 4;
  __bf16* enc_wf = (__bf16*)p; p += (size_t)1  * 4096 * 2;
  __bf16* ee_wf  = (__bf16*)p; p += (size_t)1  * 4096 * 2;
  __bf16* w1f    = (__bf16*)p; p += (size_t)3 * 4 * 4096 * 2;
  __bf16* w2f    = (__bf16*)p; p += (size_t)3 * 4 * 4096 * 2;
  __bf16* mw1f   = (__bf16*)p; p += (size_t)12 * 4096 * 2;
  float* stats   = (float*)p; p += 2 * 4;

  // --- weight preprocessing (bf16 fragment swizzle) ---
  swizzle_b_kernel<<<(1 * 4096 + 255) / 256, 256, 0, stream>>>(enc_w, enc_wf, 32, 1);
  swizzle_b_kernel<<<(1 * 4096 + 255) / 256, 256, 0, stream>>>(ee_w, ee_wf, 16, 1);
  for (int i = 0; i < 3; ++i) {
    swizzle_b_kernel<<<(4 * 4096 + 255) / 256, 256, 0, stream>>>(
        gine_w1 + (size_t)i * HID * HID, w1f + (size_t)i * 4 * 4096, 128, 4);
    swizzle_b_kernel<<<(4 * 4096 + 255) / 256, 256, 0, stream>>>(
        gine_w2 + (size_t)i * HID * HID, w2f + (size_t)i * 4 * 4096, 128, 4);
  }
  swizzle_b_kernel<<<(12 * 4096 + 255) / 256, 256, 0, stream>>>(mlp_w1, mw1f, 384, 12);

  // --- encoders ---
  enc_gemm_kernel<<<N_NODES / 16, 128, 0, stream>>>(x, 32, 32, enc_wf, enc_b, h);
  enc_gemm_kernel<<<N_EDGES / 16, 128, 0, stream>>>(eattr, 16, 16, ee_wf, ee_b, ebuf);

  // --- 3 GINE layers ---
  const size_t na = (size_t)N_NODES * HID;
  for (int i = 0; i < 3; ++i) {
    zero_kernel<<<(unsigned)((na + 255) / 256), 256, 0, stream>>>(agg, na);
    zero_kernel<<<1, 32, 0, stream>>>(stats, 2);
    message_kernel<<<(N_EDGES * 32 + 255) / 256, 256, 0, stream>>>(h, ebuf, ei, agg);
    gine_mlp_kernel<<<N_NODES / 16, 128, 0, stream>>>(
        h, agg, gine_eps, i, w1f + (size_t)i * 4 * 4096, gine_b1 + i * HID,
        w2f + (size_t)i * 4 * 4096, gine_b2 + i * HID, z, stats);
    ln_relu_kernel<<<(unsigned)((na + 255) / 256), 256, 0, stream>>>(
        z, stats, ln_g + i * HID, ln_b + i * HID, h);
  }

  // --- edge MLP head ---
  edge_mlp_kernel<<<N_EDGES / 16, 128, 0, stream>>>(h, ebuf, ei, mw1f, mlp_b1,
                                                    mlp_w2, mlp_b2, out);
}
